// MultiHeadAttention_34325378630220
// MI455X (gfx1250) — matrix operand order repaired and checked by reading, not compiled, not run on hardware
//
#include <hip/hip_runtime.h>

// Problem constants (reference: B=4, S=1024, D=1024, H=16, HD=64)
#define B_  4
#define S_  1024
#define D_  1024
#define H_  16
#define HD_ 64
#define BH_ (B_ * H_)

typedef _Float16 half_t;
typedef __attribute__((ext_vector_type(16))) _Float16 v16h;
typedef __attribute__((ext_vector_type(8)))  _Float16 v8h;
typedef __attribute__((ext_vector_type(8)))  float    v8f;
typedef __attribute__((ext_vector_type(4)))  float    v4f;
typedef __attribute__((ext_vector_type(4)))  int      v4i;

union V16U { v16h v; v8h h[2]; };

static __device__ __forceinline__ v16h pack16(v8h lo, v8h hi) {
    V16U u; u.h[0] = lo; u.h[1] = hi; return u.v;
}
static __device__ __forceinline__ v8f zero8() {
    v8f z;
#pragma unroll
    for (int i = 0; i < 8; ++i) z[i] = 0.0f;
    return z;
}

#define WMMA16(a, b, c) \
    __builtin_amdgcn_wmma_f32_16x16x32_f16(false, (a), false, (b), (short)0, (c), false, false)

// ---------------------------------------------------------------------------
// Async global->LDS copy (16B per lane), ASYNCcnt-tracked (CDNA5 path).
// Builtin signature (from compile probe): (v4i* global_src, v4i* lds_dst,
// imm offset, imm cpol); generic pointers convert implicitly under HIP's
// CUDA address-space rules.
// ---------------------------------------------------------------------------
static __device__ __forceinline__ void async_ld_b128(void* lds_dst, const void* gsrc) {
#if __has_builtin(__builtin_amdgcn_global_load_async_to_lds_b128)
    __builtin_amdgcn_global_load_async_to_lds_b128((v4i*)gsrc, (v4i*)lds_dst, 0, 0);
#else
    asm volatile("global_load_async_to_lds_b128 %0, %1, off"
                 :: "v"((unsigned)(uintptr_t)lds_dst), "v"(gsrc)
                 : "memory");
#endif
}
static __device__ __forceinline__ void async_wait_all() {
    asm volatile("s_wait_asynccnt 0" ::: "memory");
}

// ---------------------------------------------------------------------------
// Projection GEMM: P = X(fp32)[4096x1024] @ W(fp32)[1024x1024] + bias, scaled,
// written f16 head-split:
//   TRANSPOSE=0 : out[(b*H+h)][s][hd]   (Q, K)
//   TRANSPOSE=1 : out[(b*H+h)][hd][s]   (V transposed -> WMMA B layout)
// 256 thr (8 waves), tile 128x128, K-step 32, register-staged double buffer:
// next K-step's global loads are issued before the WMMA burst consumes LDS.
// ---------------------------------------------------------------------------
template <int TRANSPOSE>
__global__ __launch_bounds__(256)
void mha_gemm_proj(const float* __restrict__ X, const float* __restrict__ W,
                   const float* __restrict__ bias, half_t* __restrict__ out,
                   float outScale)
{
    __shared__ alignas(32) half_t lA[128 * 32];   // [m][k]
    __shared__ alignas(32) half_t lBt[128 * 32];  // [n][k] (W transposed)

    const int tid   = threadIdx.x;
    const int lane  = tid & 31;
    const int wave  = tid >> 5;
    const int waveM = wave >> 1;   // 0..3
    const int waveN = wave & 1;    // 0..1
    const int row0  = blockIdx.y * 128;
    const int col0  = blockIdx.x * 128;
    const int lm    = lane & 15;
    const int hs    = lane >> 4;

    // cooperative-load roles
    const int rA = tid >> 1, cA = (tid & 1) * 16;   // A: 2 thr/row, 16 floats each
    const int kW = tid >> 3, nW = (tid & 7) * 16;   // W: 8 thr/row, 16 floats each

    v4f regA[4], regW[4];
    {   // prologue: stage K-step 0
        const float* srcA = X + (size_t)(row0 + rA) * D_ + cA;
        const float* srcW = W + (size_t)kW * D_ + col0 + nW;
#pragma unroll
        for (int i = 0; i < 4; ++i) {
            regA[i] = *(const v4f*)(srcA + 4 * i);
            regW[i] = *(const v4f*)(srcW + 4 * i);
        }
    }

    v8f acc[2][4];
#pragma unroll
    for (int mt = 0; mt < 2; ++mt)
#pragma unroll
        for (int nt = 0; nt < 4; ++nt) acc[mt][nt] = zero8();

    for (int k0 = 0; k0 < D_; k0 += 32) {
        // drain staged registers -> LDS (fp32 -> f16)
        {
            half_t* dA = lA + rA * 32 + cA;
#pragma unroll
            for (int i = 0; i < 4; ++i)
#pragma unroll
                for (int j = 0; j < 4; ++j) dA[4 * i + j] = (half_t)regA[i][j];
#pragma unroll
            for (int i = 0; i < 4; ++i)
#pragma unroll
                for (int j = 0; j < 4; ++j)
                    lBt[(nW + 4 * i + j) * 32 + kW] = (half_t)regW[i][j];
        }
        __syncthreads();

        // issue next K-step's global loads early (overlap with WMMA burst)
        if (k0 + 32 < D_) {
            const float* srcA = X + (size_t)(row0 + rA) * D_ + (k0 + 32) + cA;
            const float* srcW = W + (size_t)(k0 + 32 + kW) * D_ + col0 + nW;
#pragma unroll
            for (int i = 0; i < 4; ++i) {
                regA[i] = *(const v4f*)(srcA + 4 * i);
                regW[i] = *(const v4f*)(srcW + 4 * i);
            }
        }

        // A fragments: half 0 -> k 0..15, half 1 -> k 16..31
        v16h afrag[2];
#pragma unroll
        for (int mt = 0; mt < 2; ++mt) {
            const half_t* p = lA + (waveM * 32 + mt * 16 + lm) * 32;
            afrag[mt] = pack16(*(const v8h*)(p + hs * 16),
                               *(const v8h*)(p + hs * 16 + 8));
        }
#pragma unroll
        for (int nt = 0; nt < 4; ++nt) {
            const half_t* p = lBt + (waveN * 64 + nt * 16 + lm) * 32 + hs * 16;
            v16h bfrag = *(const v16h*)p;
#pragma unroll
            for (int mt = 0; mt < 2; ++mt)
                acc[mt][nt] = WMMA16(afrag[mt], bfrag, acc[mt][nt]);
        }
        __syncthreads();
    }

    // epilogue (TRANSPOSE is compile-time: straight-line stores)
#pragma unroll
    for (int nt = 0; nt < 4; ++nt) {
        const int n  = col0 + waveN * 64 + nt * 16 + lm;
        const float bv = bias[n];
        const int h  = n >> 6;
        const int hd = n & 63;
#pragma unroll
        for (int mt = 0; mt < 2; ++mt) {
#pragma unroll
            for (int r = 0; r < 8; ++r) {
                const int m  = row0 + waveM * 32 + mt * 16 + hs * 8 + r;
                const int bb = m >> 10;
                const int s  = m & (S_ - 1);
                const int bh = bb * H_ + h;
                const float val = (acc[mt][nt][r] + bv) * outScale;
                if (TRANSPOSE)
                    out[(size_t)(bh * HD_ + hd) * S_ + s] = (half_t)val;
                else
                    out[(size_t)(bh * S_ + s) * HD_ + hd] = (half_t)val;
            }
        }
    }
}

// ---------------------------------------------------------------------------
// Fused flash attention: block = one (bh, 64-row q-tile), 4 waves x 16 rows.
// K/V^T/mask tiles stream through double-buffered LDS via async-to-LDS copies
// (ASYNCcnt), overlapping the next tile's HBM traffic with the current tile's
// WMMA + softmax. Score tensor never touches HBM. Mask uses bh % 4 to
// reproduce the reference's reshape(H,B,S,S) quirk.
// ---------------------------------------------------------------------------
__global__ __launch_bounds__(128)
void mha_flash_attn(const half_t* __restrict__ Qh,   // [bh][s][hd], pre-scaled 1/8
                    const half_t* __restrict__ Kh,   // [bh][s][hd]
                    const half_t* __restrict__ Vth,  // [bh][hd][s]
                    const int*    __restrict__ mask, // [4][S][S]
                    half_t* __restrict__ Oc)         // [B][S][D] f16
{
    __shared__ alignas(32) half_t lK [2][64 * 64];   // [kk][d]
    __shared__ alignas(32) half_t lVt[2][64 * 64];   // [hd][kk]
    __shared__ alignas(16) int    lMask[2][64 * 64]; // [q][kk]
    __shared__ alignas(32) half_t lP [4 * 16 * 64];  // per-wave P round-trip

    const int bh  = blockIdx.y;
    const int q0  = blockIdx.x * 64;
    const int mb  = bh & (B_ - 1);
    const int tid = threadIdx.x;
    const int lane = tid & 31;
    const int wave = tid >> 5;
    const int lm   = lane & 15;
    const int hs   = lane >> 4;

    // per-thread async copy roles (16B granules)
    const int cpRow = tid >> 1;              // 0..63
    const int cpC32 = (tid & 1) * 32;        // half_t offset (64B half)

    auto issue_tile = [&](int k0, int buf) {
        {   // K tile [kk][d]
            const half_t* src = Kh + ((size_t)bh * S_ + k0 + cpRow) * HD_ + cpC32;
            half_t* dst = &lK[buf][cpRow * 64 + cpC32];
#pragma unroll
            for (int i = 0; i < 4; ++i) async_ld_b128(dst + i * 8, src + i * 8);
        }
        {   // V^T tile [hd][kk]
            const half_t* src = Vth + ((size_t)(bh * HD_ + cpRow)) * S_ + k0 + cpC32;
            half_t* dst = &lVt[buf][cpRow * 64 + cpC32];
#pragma unroll
            for (int i = 0; i < 4; ++i) async_ld_b128(dst + i * 8, src + i * 8);
        }
        {   // mask tile [q][kk]
            const int* src = mask + ((size_t)mb * S_ + q0 + cpRow) * S_ + k0 + cpC32;
            int* dst = &lMask[buf][cpRow * 64 + cpC32];
#pragma unroll
            for (int i = 0; i < 8; ++i) async_ld_b128(dst + i * 4, src + i * 4);
        }
    };

    // Q fragments for this wave's 16 rows (d = 64 -> 2 k-steps of 32)
    const half_t* qp = Qh + ((size_t)bh * S_ + q0 + wave * 16 + lm) * HD_;
    v16h aq[2];
#pragma unroll
    for (int ks = 0; ks < 2; ++ks)
        aq[ks] = pack16(*(const v8h*)(qp + ks * 32 + hs * 16),
                        *(const v8h*)(qp + ks * 32 + hs * 16 + 8));

    v8f oacc[4];
#pragma unroll
    for (int ht = 0; ht < 4; ++ht) oacc[ht] = zero8();
    float mrow[8], lrow[8];
#pragma unroll
    for (int r = 0; r < 8; ++r) { mrow[r] = -1e30f; lrow[r] = 0.0f; }

    half_t* pw = lP + wave * (16 * 64);

    issue_tile(0, 0);
    async_wait_all();
    __syncthreads();

    for (int t = 0; t < S_ / 64; ++t) {
        const int buf = t & 1;
        if (t + 1 < S_ / 64) issue_tile((t + 1) * 64, buf ^ 1);  // overlap copy w/ compute

        const half_t* Kt = &lK[buf][0];
        const half_t* Vt = &lVt[buf][0];
        const int*    Mt = &lMask[buf][0];

        // ---- scores: S = Q K^T (1/sqrt(HD) folded into Q) ----
        v8f sacc[4];
#pragma unroll
        for (int nt = 0; nt < 4; ++nt) {
            v8f c = zero8();
#pragma unroll
            for (int ks = 0; ks < 2; ++ks) {
                const half_t* bp = Kt + (nt * 16 + lm) * 64 + ks * 32 + hs * 16;
                v16h bk = *(const v16h*)bp;
                c = WMMA16(aq[ks], bk, c);
            }
            sacc[nt] = c;
        }

        // ---- mask + online softmax ----
        float tmax[8];
#pragma unroll
        for (int r = 0; r < 8; ++r) tmax[r] = -1e30f;
#pragma unroll
        for (int nt = 0; nt < 4; ++nt) {
            const int kcol = nt * 16 + lm;
#pragma unroll
            for (int r = 0; r < 8; ++r) {
                const int qrow = wave * 16 + hs * 8 + r;
                float sv = sacc[nt][r];
                if (Mt[qrow * 64 + kcol] == 0) sv = -1e9f;
                sacc[nt][r] = sv;
                tmax[r] = fmaxf(tmax[r], sv);
            }
        }
#pragma unroll
        for (int off = 1; off < 16; off <<= 1)
#pragma unroll
            for (int r = 0; r < 8; ++r)
                tmax[r] = fmaxf(tmax[r], __shfl_xor(tmax[r], off, 32));

        float alpha[8];
#pragma unroll
        for (int r = 0; r < 8; ++r) {
            const float mn = fmaxf(mrow[r], tmax[r]);
            alpha[r] = __expf(mrow[r] - mn);
            mrow[r] = mn;
        }
        float tsum[8];
#pragma unroll
        for (int r = 0; r < 8; ++r) tsum[r] = 0.0f;
#pragma unroll
        for (int nt = 0; nt < 4; ++nt)
#pragma unroll
            for (int r = 0; r < 8; ++r) {
                const float p = __expf(sacc[nt][r] - mrow[r]);
                sacc[nt][r] = p;
                tsum[r] += p;
            }
#pragma unroll
        for (int off = 1; off < 16; off <<= 1)
#pragma unroll
            for (int r = 0; r < 8; ++r)
                tsum[r] += __shfl_xor(tsum[r], off, 32);
#pragma unroll
        for (int r = 0; r < 8; ++r) lrow[r] = lrow[r] * alpha[r] + tsum[r];
#pragma unroll
        for (int ht = 0; ht < 4; ++ht)
#pragma unroll
            for (int r = 0; r < 8; ++r) oacc[ht][r] *= alpha[r];

        // ---- C-layout -> A-layout via per-wave LDS round trip ----
#pragma unroll
        for (int nt = 0; nt < 4; ++nt) {
            const int kcol = nt * 16 + lm;
#pragma unroll
            for (int r = 0; r < 8; ++r)
                pw[(hs * 8 + r) * 64 + kcol] = (half_t)sacc[nt][r];
        }
        v16h ap[2];
#pragma unroll
        for (int ks = 0; ks < 2; ++ks) {
            const half_t* pp = pw + lm * 64 + ks * 32;
            ap[ks] = pack16(*(const v8h*)(pp + hs * 16),
                            *(const v8h*)(pp + hs * 16 + 8));
        }

        // ---- O += P @ V ----
#pragma unroll
        for (int ht = 0; ht < 4; ++ht) {
#pragma unroll
            for (int ks = 0; ks < 2; ++ks) {
                const half_t* bp = Vt + (ht * 16 + lm) * 64 + ks * 32 + hs * 16;
                v16h bv = *(const v16h*)bp;
                oacc[ht] = WMMA16(ap[ks], bv, oacc[ht]);
            }
        }

        async_wait_all();   // next tile's async copies landed in LDS
        __syncthreads();
    }

    // ---- normalize, store f16 into concat [B][S][D] ----
    const int h  = bh & (H_ - 1);
    const int bb = bh >> 4;
#pragma unroll
    for (int ht = 0; ht < 4; ++ht) {
        const int hd = ht * 16 + lm;
        const int d  = h * HD_ + hd;
#pragma unroll
        for (int r = 0; r < 8; ++r) {
            const int s = q0 + wave * 16 + hs * 8 + r;
            const float val = oacc[ht][r] / lrow[r];
            Oc[((size_t)(bb * S_ + s)) * D_ + d] = (half_t)val;
        }
    }
}

// ---------------------------------------------------------------------------
// Output GEMM: out(fp32) = Oc(f16)[4096x1024] @ Wo(fp32) + bo
// Register-staged double buffer like the projection GEMM.
// ---------------------------------------------------------------------------
__global__ __launch_bounds__(256)
void mha_gemm_out(const half_t* __restrict__ A, const float* __restrict__ W,
                  const float* __restrict__ bias, float* __restrict__ out)
{
    __shared__ alignas(32) half_t lA[128 * 32];
    __shared__ alignas(32) half_t lBt[128 * 32];

    const int tid   = threadIdx.x;
    const int lane  = tid & 31;
    const int wave  = tid >> 5;
    const int waveM = wave >> 1;
    const int waveN = wave & 1;
    const int row0  = blockIdx.y * 128;
    const int col0  = blockIdx.x * 128;
    const int lm    = lane & 15;
    const int hs    = lane >> 4;

    const int rA = tid >> 1, cA = (tid & 1) * 16;
    const int kW = tid >> 3, nW = (tid & 7) * 16;

    v8h regA[2];
    v4f regW[4];
    {
        const half_t* srcA = A + (size_t)(row0 + rA) * D_ + cA;
        const float*  srcW = W + (size_t)kW * D_ + col0 + nW;
        regA[0] = *(const v8h*)(srcA);
        regA[1] = *(const v8h*)(srcA + 8);
#pragma unroll
        for (int i = 0; i < 4; ++i) regW[i] = *(const v4f*)(srcW + 4 * i);
    }

    v8f acc[2][4];
#pragma unroll
    for (int mt = 0; mt < 2; ++mt)
#pragma unroll
        for (int nt = 0; nt < 4; ++nt) acc[mt][nt] = zero8();

    for (int k0 = 0; k0 < D_; k0 += 32) {
        {
            half_t* dA = lA + rA * 32 + cA;
            *(v8h*)(dA)     = regA[0];
            *(v8h*)(dA + 8) = regA[1];
#pragma unroll
            for (int i = 0; i < 4; ++i)
#pragma unroll
                for (int j = 0; j < 4; ++j)
                    lBt[(nW + 4 * i + j) * 32 + kW] = (half_t)regW[i][j];
        }
        __syncthreads();

        if (k0 + 32 < D_) {
            const half_t* srcA = A + (size_t)(row0 + rA) * D_ + (k0 + 32) + cA;
            const float*  srcW = W + (size_t)(k0 + 32 + kW) * D_ + col0 + nW;
            regA[0] = *(const v8h*)(srcA);
            regA[1] = *(const v8h*)(srcA + 8);
#pragma unroll
            for (int i = 0; i < 4; ++i) regW[i] = *(const v4f*)(srcW + 4 * i);
        }

        v16h afrag[2];
#pragma unroll
        for (int mt = 0; mt < 2; ++mt) {
            const half_t* p = lA + (waveM * 32 + mt * 16 + lm) * 32;
            afrag[mt] = pack16(*(const v8h*)(p + hs * 16),
                               *(const v8h*)(p + hs * 16 + 8));
        }
#pragma unroll
        for (int nt = 0; nt < 4; ++nt) {
            const half_t* p = lBt + (waveN * 64 + nt * 16 + lm) * 32 + hs * 16;
            v16h bfrag = *(const v16h*)p;
#pragma unroll
            for (int mt = 0; mt < 2; ++mt)
                acc[mt][nt] = WMMA16(afrag[mt], bfrag, acc[mt][nt]);
        }
        __syncthreads();
    }

#pragma unroll
    for (int nt = 0; nt < 4; ++nt) {
        const int n = col0 + waveN * 64 + nt * 16 + lm;
        const float bv = bias[n];
#pragma unroll
        for (int mt = 0; mt < 2; ++mt) {
#pragma unroll
            for (int r = 0; r < 8; ++r) {
                const int m = row0 + waveM * 32 + mt * 16 + hs * 8 + r;
                out[(size_t)m * D_ + n] = acc[mt][nt][r] + bv;
            }
        }
    }
}

// ---------------------------------------------------------------------------
// Harness entry
// ---------------------------------------------------------------------------
extern "C" void kernel_launch(void* const* d_in, const int* in_sizes, int n_in,
                              void* d_out, int out_size, void* d_ws, size_t ws_size,
                              hipStream_t stream) {
    (void)in_sizes; (void)n_in; (void)out_size; (void)ws_size;

    const float* query = (const float*)d_in[0];
    const float* key_  = (const float*)d_in[1];
    const float* value = (const float*)d_in[2];
    const int*   mask  = (const int*)d_in[3];
    const float* Wq = (const float*)d_in[4];
    const float* bq = (const float*)d_in[5];
    const float* Wk = (const float*)d_in[6];
    const float* bk = (const float*)d_in[7];
    const float* Wv = (const float*)d_in[8];
    const float* bv = (const float*)d_in[9];
    const float* Wo = (const float*)d_in[10];
    const float* bo = (const float*)d_in[11];

    // workspace: 4 x 8 MB f16 buffers = 32 MB
    half_t* Qh  = (half_t*)d_ws;                       // [bh][s][hd], pre-scaled 1/8
    half_t* Kh  = Qh  + (size_t)BH_ * S_ * HD_;        // [bh][s][hd]
    half_t* Vth = Kh  + (size_t)BH_ * S_ * HD_;        // [bh][hd][s] (transposed)
    half_t* Oc  = Vth + (size_t)BH_ * S_ * HD_;        // [B][S][D] attention output

    dim3 gg(D_ / 128, (B_ * S_) / 128);                // (8, 32)
    mha_gemm_proj<0><<<gg, 256, 0, stream>>>(query, Wq, bq, Qh, 0.125f); // 1/sqrt(64)
    mha_gemm_proj<0><<<gg, 256, 0, stream>>>(key_,  Wk, bk, Kh, 1.0f);
    mha_gemm_proj<1><<<gg, 256, 0, stream>>>(value, Wv, bv, Vth, 1.0f);

    mha_flash_attn<<<dim3(S_ / 64, BH_), 128, 0, stream>>>(Qh, Kh, Vth, mask, Oc);

    mha_gemm_out<<<gg, 256, 0, stream>>>(Oc, Wo, bo, (float*)d_out);
}
